// MultiQueryAttention_7919919694266
// MI455X (gfx1250) — matrix operand order repaired and checked by reading, not compiled, not run on hardware
//
#include <hip/hip_runtime.h>

#define BB 4
#define TT 2048
#define EE 1024
#define HH 8
#define DD 128
#define MM (BB*TT)   // 8192 rows

typedef __attribute__((ext_vector_type(16))) __bf16 bf16x16;
typedef __attribute__((ext_vector_type(8)))  float  f32x8;
typedef __attribute__((ext_vector_type(4)))  float  f32x4;
typedef __attribute__((ext_vector_type(4)))  unsigned int u32x4;
typedef __attribute__((ext_vector_type(2)))  unsigned int u32x2;

union Frag {
  bf16x16 v;
  u32x4   q[2];
  unsigned short s[16];
};

__device__ __forceinline__ unsigned short f2bf(float f) {
  unsigned int u = __float_as_uint(f);
  u += 0x7FFFu + ((u >> 16) & 1u);      // round-to-nearest-even
  return (unsigned short)(u >> 16);
}

__device__ __forceinline__ unsigned int pk2bf(float lo, float hi) {
  return (unsigned int)f2bf(lo) | ((unsigned int)f2bf(hi) << 16);
}

__device__ __forceinline__ f32x8 wmma_bf16(Frag a, Frag b, f32x8 c) {
  return __builtin_amdgcn_wmma_f32_16x16x32_bf16(
      false, a.v, false, b.v, (short)0, c, false, false);
}

__device__ __forceinline__ Frag ldfrag(const unsigned short* base, unsigned eo) {
  Frag f;
  f.q[0] = *(const u32x4*)(base + eo);
  f.q[1] = *(const u32x4*)(base + eo + 8);
  return f;
}

// ---------------------------------------------------------------------------
// fp32 -> bf16 conversion, 4 elements/lane
// ---------------------------------------------------------------------------
__global__ void cvt_bf16_kernel(const float* __restrict__ in,
                                unsigned short* __restrict__ out, int n) {
  int i = (blockIdx.x * blockDim.x + threadIdx.x) * 4;
  if (i < n) {
    f32x4 v = *(const f32x4*)(in + i);
    u32x2 p;
    p[0] = pk2bf(v[0], v[1]);
    p[1] = pk2bf(v[2], v[3]);
    *(u32x2*)(out + i) = p;
  }
}

// ---------------------------------------------------------------------------
// Y(M,N) = Xbf(M,K) @ Wbf(N,K)^T, fp32 accumulate. 32x64 tile per wave
// (2 M-subtiles x 4 N-subtiles => 8 WMMA per k-step with A+B reuse),
// register double-buffered k-loop (peeled tail, no cselect), 32-bit element
// offsets from uniform bases, WGP-scope prefetch of future k-chunks.
// mode 0: write bf16
// mode 1: Y = (Y + PE(t=m%T, d=n%D)) * scale, write bf16              (Q)
// mode 2: Y += (n&1) ? 1.0 : 0.0, write bf16                          (K)
// mode 3: write fp32                                                   (out)
// mode 4: write bf16 TRANSPOSED: Y[n*M + m] (contiguous 16B/lane)      (V)
// ---------------------------------------------------------------------------
__global__ void gemm_bf16_kernel(const unsigned short* __restrict__ X,
                                 const unsigned short* __restrict__ W,
                                 void* __restrict__ Y,
                                 int M, int N, int K, int nshift,
                                 int mode, float scale) {
  const int lane = threadIdx.x & 31;
  const int half = lane >> 4;
  const int lr   = lane & 15;
  const int wid  = blockIdx.x * (blockDim.x >> 5) + (threadIdx.x >> 5);
  const int tm   = wid >> nshift;
  const int tn   = wid & ((1 << nshift) - 1);

  f32x8 acc[2][4];
#pragma unroll
  for (int s = 0; s < 2; ++s)
#pragma unroll
    for (int j = 0; j < 4; ++j) acc[s][j] = (f32x8){0,0,0,0,0,0,0,0};

  // 32-bit element offsets from uniform bases (saddr + voffset addressing)
  unsigned xo[2], wo[4];
#pragma unroll
  for (int s = 0; s < 2; ++s)
    xo[s] = (unsigned)(tm * 32 + s * 16 + lr) * (unsigned)K + 16u * half;
#pragma unroll
  for (int j = 0; j < 4; ++j)
    wo[j] = (unsigned)(tn * 64 + j * 16 + lr) * (unsigned)K + 16u * half;

  // prologue: k-chunk 0
  Frag a0[2], b0[4];
#pragma unroll
  for (int s = 0; s < 2; ++s) a0[s] = ldfrag(X, xo[s]);
#pragma unroll
  for (int j = 0; j < 4; ++j) b0[j] = ldfrag(W, wo[j]);

  const int steps = K >> 5;
  for (int it = 0; it < steps - 1; ++it) {
    Frag a1[2], b1[4];
#pragma unroll
    for (int s = 0; s < 2; ++s) {
      xo[s] += 32u;
      a1[s] = ldfrag(X, xo[s]);
      __builtin_prefetch(X + xo[s] + 128u, 0, 3);   // WGP-scope prefetch
    }
#pragma unroll
    for (int j = 0; j < 4; ++j) {
      wo[j] += 32u;
      b1[j] = ldfrag(W, wo[j]);
      __builtin_prefetch(W + wo[j] + 128u, 0, 3);
    }
#pragma unroll
    for (int j = 0; j < 4; ++j)
#pragma unroll
      for (int s = 0; s < 2; ++s)
        acc[s][j] = wmma_bf16(a0[s], b0[j], acc[s][j]);
#pragma unroll
    for (int s = 0; s < 2; ++s) a0[s] = a1[s];
#pragma unroll
    for (int j = 0; j < 4; ++j) b0[j] = b1[j];
  }
#pragma unroll
  for (int j = 0; j < 4; ++j)
#pragma unroll
    for (int s = 0; s < 2; ++s)
      acc[s][j] = wmma_bf16(a0[s], b0[j], acc[s][j]);

  if (mode == 4) {
    // transposed store: rows m are contiguous per lane -> one 16B store each
#pragma unroll
    for (int s = 0; s < 2; ++s)
#pragma unroll
      for (int j = 0; j < 4; ++j) {
        u32x4 pk;
#pragma unroll
        for (int rr = 0; rr < 4; ++rr)
          pk[rr] = pk2bf(acc[s][j][2 * rr], acc[s][j][2 * rr + 1]);
        int n = tn * 64 + j * 16 + lr;
        *(u32x4*)((unsigned short*)Y +
                  (size_t)n * M + tm * 32 + s * 16 + 8 * half) = pk;
      }
    return;
  }

#pragma unroll
  for (int s = 0; s < 2; ++s)
#pragma unroll
    for (int j = 0; j < 4; ++j)
#pragma unroll
      for (int r = 0; r < 8; ++r) {
        int m = tm * 32 + s * 16 + r + 8 * half;
        int n = tn * 64 + j * 16 + lr;
        float v = acc[s][j][r];
        if (mode == 1) {
          int t = m & (TT - 1);
          int d = n & (DD - 1);
          float fr  = __expf(-9.210340371976184f *
                             ((float)(d & ~1) / (float)DD));
          float ang = (float)t * fr;
          v = (v + ((d & 1) ? __cosf(ang) : __sinf(ang))) * scale;
        } else if (mode == 2) {
          v += (n & 1) ? 1.0f : 0.0f;
        }
        if (mode == 3)
          ((float*)Y)[(size_t)m * N + n] = v;
        else
          ((unsigned short*)Y)[(size_t)m * N + n] = f2bf(v);
      }
}

// ---------------------------------------------------------------------------
// Flash attention, one wave per 16-query tile of one (b,h).
// Scores computed transposed (S^T = K . Q^T): softmax key-reduction is
// lane-local (+ one shfl_xor 16); P^T C-tiles repack into the P A-fragment
// lane-locally. V is pre-transposed (Vt[d, b*T+t]) so the V B-fragment is
// two contiguous b128 loads per 16-wide d-chunk.
// ---------------------------------------------------------------------------
__global__ void attn_kernel(const unsigned short* __restrict__ Q,
                            const unsigned short* __restrict__ Kb,
                            const unsigned short* __restrict__ Vt,
                            unsigned short* __restrict__ O) {
  const int lane = threadIdx.x & 31;
  const int half = lane >> 4;
  const int lr   = lane & 15;
  const int wid  = blockIdx.x * (blockDim.x >> 5) + (threadIdx.x >> 5);
  const int qtiles = TT / 16;
  const int qt = wid % qtiles;
  const int bh = wid / qtiles;
  const int b  = bh / HH;
  const int h  = bh % HH;
  const int q0 = qt * 16;

  // Q^T as B-fragments: lane n = query, halves = contiguous d.
  const unsigned qoff =
      (unsigned)(b * TT + q0 + lr) * (HH * DD) + (unsigned)(h * DD);
  Frag qf[4];
#pragma unroll
  for (int c = 0; c < 4; ++c)
    qf[c] = ldfrag(Q, qoff + c * 32u + 16u * half);

  f32x8 o[8];
#pragma unroll
  for (int n = 0; n < 8; ++n) o[n] = (f32x8){0,0,0,0,0,0,0,0};
  float mprev = -1e30f, lsum = 0.0f;

  const int tq  = q0 + lr;          // this lane's query index
  const int nkt = (q0 + 47) >> 5;   // 32-key tiles covering [0, q0+15]

  // running 32-bit offsets
  unsigned ko = (unsigned)(b * TT + lr) * DD + 16u * half;       // K rows
  unsigned vo = (unsigned)lr * MM + (unsigned)(b * TT) + 8u * half;  // Vt

  for (int kt = 0; kt < nkt; ++kt) {
    const int s0 = kt * 32;
    f32x8 c0 = (f32x8){0,0,0,0,0,0,0,0};
    f32x8 c1 = (f32x8){0,0,0,0,0,0,0,0};
    __builtin_prefetch(Kb + ko + 32u * DD, 0, 3);
    __builtin_prefetch(Kb + ko + 48u * DD, 0, 3);
#pragma unroll
    for (int c = 0; c < 4; ++c) {
      Frag a0 = ldfrag(Kb, ko + c * 32u);
      c0 = wmma_bf16(a0, qf[c], c0);
      Frag a1 = ldfrag(Kb, ko + 16u * DD + c * 32u);
      c1 = wmma_bf16(a1, qf[c], c1);
    }
    ko += 32u * DD;

    // causal mask (rows m of C = keys, column n = lane's query)
    if (s0 + 31 > q0) {
#pragma unroll
      for (int r = 0; r < 8; ++r) {
        int s = s0 + r + 8 * half;
        if (s > tq)      c0[r] = -1e30f;
        if (s + 16 > tq) c1[r] = -1e30f;
      }
    }

    // online softmax (per-lane over keys, combined across lane halves)
    float tmax = -1e30f;
#pragma unroll
    for (int r = 0; r < 8; ++r) tmax = fmaxf(tmax, fmaxf(c0[r], c1[r]));
    tmax = fmaxf(tmax, __shfl_xor(tmax, 16, 32));
    float mnew  = fmaxf(mprev, tmax);
    float alpha = __expf(mprev - mnew);
    float p0[8], p1[8], rs = 0.0f;
#pragma unroll
    for (int r = 0; r < 8; ++r) {
      p0[r] = __expf(c0[r] - mnew);
      p1[r] = __expf(c1[r] - mnew);
      rs += p0[r] + p1[r];
    }
    rs += __shfl_xor(rs, 16, 32);
    lsum = lsum * alpha + rs;
    mprev = mnew;

    // repack P^T (C layout) into P A-fragment: lane-local
    Frag pa;
#pragma unroll
    for (int r = 0; r < 8; ++r) {
      pa.s[r]     = f2bf(p0[r]);
      pa.s[8 + r] = f2bf(p1[r]);
    }

    // per-row rescale factors for O (rows = queries)
    float arow[8];
#pragma unroll
    for (int r = 0; r < 8; ++r) arow[r] = __shfl(alpha, r + 8 * half, 32);

    unsigned vn = vo;
#pragma unroll
    for (int n = 0; n < 8; ++n) {
      f32x8 oo = o[n];
#pragma unroll
      for (int r = 0; r < 8; ++r) oo[r] *= arow[r];
      // V B-frag from transposed V: contiguous keys along rows of Vt
      __builtin_prefetch(Vt + vn + 32u, 0, 3);
      Frag bv;
      bv.q[0] = *(const u32x4*)(Vt + vn);
      bv.q[1] = *(const u32x4*)(Vt + vn + 16u);
      o[n] = wmma_bf16(pa, bv, oo);
      vn += 16u * MM;
    }
    vo += 32u;
  }

  float linv = 1.0f / lsum;
  float lrow[8];
#pragma unroll
  for (int r = 0; r < 8; ++r) lrow[r] = __shfl(linv, r + 8 * half, 32);
#pragma unroll
  for (int n = 0; n < 8; ++n) {
#pragma unroll
    for (int r = 0; r < 8; ++r) {
      size_t row = (size_t)(b * TT + q0 + r + 8 * half);
      O[row * (HH * DD) + h * DD + n * 16 + lr] = f2bf(o[n][r] * lrow[r]);
    }
  }
}

// ---------------------------------------------------------------------------
extern "C" void kernel_launch(void* const* d_in, const int* in_sizes, int n_in,
                              void* d_out, int out_size, void* d_ws, size_t ws_size,
                              hipStream_t stream) {
  (void)in_sizes; (void)n_in; (void)out_size; (void)ws_size;
  const float* x  = (const float*)d_in[0];
  const float* Wq = (const float*)d_in[1];
  const float* Wk = (const float*)d_in[2];
  const float* Wv = (const float*)d_in[3];
  const float* Wo = (const float*)d_in[4];

  char* ws = (char*)d_ws;
  unsigned short* xbf = (unsigned short*)ws;              ws += (size_t)MM * EE * 2;
  unsigned short* wqb = (unsigned short*)ws;              ws += (size_t)(HH*DD) * EE * 2;
  unsigned short* wkb = (unsigned short*)ws;              ws += (size_t)DD * EE * 2;
  unsigned short* wvb = (unsigned short*)ws;              ws += (size_t)DD * EE * 2;
  unsigned short* wob = (unsigned short*)ws;              ws += (size_t)EE * (HH*DD) * 2;
  unsigned short* Qb  = (unsigned short*)ws;              ws += (size_t)MM * (HH*DD) * 2;
  unsigned short* Kb  = (unsigned short*)ws;              ws += (size_t)MM * DD * 2;
  unsigned short* Vtb = (unsigned short*)ws;              ws += (size_t)MM * DD * 2;
  unsigned short* Ab  = (unsigned short*)ws;

  const int nx  = MM * EE;
  const int nwq = HH * DD * EE;
  const int nwk = DD * EE;
  cvt_bf16_kernel<<<nx  / 1024, 256, 0, stream>>>(x,  xbf, nx);
  cvt_bf16_kernel<<<nwq / 1024, 256, 0, stream>>>(Wq, wqb, nwq);
  cvt_bf16_kernel<<<nwk / 1024, 256, 0, stream>>>(Wk, wkb, nwk);
  cvt_bf16_kernel<<<nwk / 1024, 256, 0, stream>>>(Wv, wvb, nwk);
  cvt_bf16_kernel<<<nwq / 1024, 256, 0, stream>>>(Wo, wob, nwq);

  const float scale = 0.08838834764831845f;  // 1/sqrt(128)

  // Q = (x@Wq^T + PE(t,d)) * scale -> bf16, M=8192 N=1024 K=1024
  gemm_bf16_kernel<<<512, 256, 0, stream>>>(xbf, wqb, Qb,
      MM, HH * DD, EE, 4, 1, scale);
  // K = x@Wk^T + pe(1,D) -> bf16, N=128
  gemm_bf16_kernel<<<64, 256, 0, stream>>>(xbf, wkb, Kb,
      MM, DD, EE, 1, 2, 1.0f);
  // V = x@Wv^T -> bf16 TRANSPOSED Vt(D, M)
  gemm_bf16_kernel<<<64, 256, 0, stream>>>(xbf, wvb, Vtb,
      MM, DD, EE, 1, 4, 1.0f);

  // flash attention: B*H*(T/16) waves / 8 per block = 512 blocks
  attn_kernel<<<(BB * HH * (TT / 16)) / 8, 256, 0, stream>>>(Qb, Kb, Vtb, Ab);

  // out = attn @ Wo^T -> fp32, M=8192 N=1024 K=1024
  gemm_bf16_kernel<<<512, 256, 0, stream>>>(Ab, wob, (float*)d_out,
      MM, EE, HH * DD, 4, 3, 1.0f);
}
